// DualChannel_82583631167768
// MI455X (gfx1250) — matrix operand order repaired and checked by reading, not compiled, not run on hardware
//
#include <hip/hip_runtime.h>
#include <math.h>

#define NN   50000     // nodes
#define NE   800000    // edges (= 16 * 50000, exact)
#define NF   256       // input features
#define HID  64        // hidden
#define NCLS 40        // classes
#define NPAD 48        // classes padded to 3 WMMA n-tiles
#define EPSF 0.3f
#define NH   (NN * HID)

typedef __attribute__((ext_vector_type(16))) _Float16 v16h;
typedef __attribute__((ext_vector_type(8)))  _Float16 v8h;
typedef __attribute__((ext_vector_type(8)))  float    v8f;

// ---------------------------------------------------------------- helpers
__device__ __forceinline__ v16h pack16(float4 a, float4 b, float4 c, float4 d) {
    v16h r;
    r[0]  = (_Float16)a.x; r[1]  = (_Float16)a.y; r[2]  = (_Float16)a.z; r[3]  = (_Float16)a.w;
    r[4]  = (_Float16)b.x; r[5]  = (_Float16)b.y; r[6]  = (_Float16)b.z; r[7]  = (_Float16)b.w;
    r[8]  = (_Float16)c.x; r[9]  = (_Float16)c.y; r[10] = (_Float16)c.z; r[11] = (_Float16)c.w;
    r[12] = (_Float16)d.x; r[13] = (_Float16)d.y; r[14] = (_Float16)d.z; r[15] = (_Float16)d.w;
    return r;
}

__device__ __forceinline__ v8h pack8(float4 a, float4 b) {
    v8h r;
    r[0] = (_Float16)a.x; r[1] = (_Float16)a.y; r[2] = (_Float16)a.z; r[3] = (_Float16)a.w;
    r[4] = (_Float16)b.x; r[5] = (_Float16)b.y; r[6] = (_Float16)b.z; r[7] = (_Float16)b.w;
    return r;
}

// ---------------------------------------------------------------- fill
__global__ void fill_zero(float* __restrict__ p, int n) {
    int i = blockIdx.x * blockDim.x + threadIdx.x;
    int stride = gridDim.x * blockDim.x;
    for (; i < n; i += stride) p[i] = 0.0f;
}

// ---------------------------------------------------------------- degrees (3125 blocks x 256, exact)
__global__ void degree_kernel(const int* __restrict__ ei,
                              float* __restrict__ deg, float* __restrict__ cnt) {
    int e = blockIdx.x * blockDim.x + threadIdx.x;
    atomicAdd(deg + ei[e],      1.0f);   // row (source) degree
    atomicAdd(cnt + ei[NE + e], 1.0f);   // col (target) in-count
}

__global__ void node_norm(float* __restrict__ nd, float* __restrict__ cnt) {
    int i = blockIdx.x * blockDim.x + threadIdx.x;
    if (i >= NN) return;
    nd[i]  = rsqrtf(fmaxf(nd[i], 1.0f)); // deg^{-1/2}, clamp >= 1
    cnt[i] = fmaxf(cnt[i], 1.0f);        // mean denominator, clamp >= 1
}

// ---------------------------------------------------------------- h = relu(x @ W1^T + b1)
// W1 is converted to f16 in LDS once per block (32 KB of the 320 KB WGP LDS);
// each wave computes a 16-row x 64-col strip: 4 accumulators of
// v_wmma_f32_16x16x32_f16, B fragments come straight from LDS (2x ds_load_b128).
__global__ __launch_bounds__(256) void proj_wmma(const float* __restrict__ x,
                                                 const float* __restrict__ W1,
                                                 const float* __restrict__ b1,
                                                 float* __restrict__ raw,
                                                 float* __restrict__ h) {
    __shared__ alignas(32) _Float16 sW[HID * NF];   // W1 as f16, row-major [64][256]

    // cooperative f32->f16 staging: 16384 elems, 8 per iteration
    for (int i = threadIdx.x; i < (HID * NF) / 8; i += 256) {
        const float* src = W1 + i * 8;
        *(v8h*)(sW + i * 8) = pack8(*(const float4*)(src), *(const float4*)(src + 4));
    }
    __syncthreads();

    const int lane    = threadIdx.x & 31;
    const int wid     = threadIdx.x >> 5;
    const int rowBase = blockIdx.x * 128 + wid * 16;
    if (rowBase >= NN) return;
    const int laneLo = lane & 15;
    const int laneHi = lane >> 4;

    const float* arow = x + (size_t)(rowBase + laneLo) * NF;
    v8f acc[4] = {};

    for (int k0 = 0; k0 < NF; k0 += 32) {
        // A tile 16x32 f16: lanes<16 hold K[0..15]; lanes>=16 K[16..31]
        const int kA = k0 + laneHi * 16;
        if (k0 + 32 < NF) __builtin_prefetch(arow + kA + 32, 0, 3);  // global_prefetch_b8
        v16h A = pack16(*(const float4*)(arow + kA),
                        *(const float4*)(arow + kA + 4),
                        *(const float4*)(arow + kA + 8),
                        *(const float4*)(arow + kA + 12));

        // B tile 32x16 f16 from LDS: lane holds column n, 16 contiguous K values
        const int kB = k0 + laneHi * 16;
#pragma unroll
        for (int t = 0; t < 4; ++t) {
            v16h B = *(const v16h*)(sW + (t * 16 + laneLo) * NF + kB);  // B[k][n] = W1[n][k]
            acc[t] = __builtin_amdgcn_wmma_f32_16x16x32_f16(
                false, A, false, B, (short)0, acc[t], false, false);
        }
    }

#pragma unroll
    for (int t = 0; t < 4; ++t) {
        const int n = t * 16 + laneLo;
        const float bias = b1[n];
#pragma unroll
        for (int v = 0; v < 8; ++v) {
            const int m = rowBase + v + 8 * laneHi;   // C/D layout: lanes<16 M=v, lanes>=16 M=8+v
            const float val = fmaxf(acc[t][v] + bias, 0.0f);
            raw[(size_t)m * HID + n] = val;
            h[(size_t)m * HID + n]   = val;
        }
    }
}

// ---------------------------------------------------------------- edge gather / gate / scatter-add
// One half-wave (16 lanes) per edge, each lane owns 4 of 64 features:
// row gathers are two coalesced b128 per edge; 800000 = 16*50000 -> no guards.
__global__ __launch_bounds__(256) void edge_gate_scatter(const float* __restrict__ h,
                                                         const int* __restrict__ ei,
                                                         const float* __restrict__ nd,
                                                         const float* __restrict__ gate_w,
                                                         const float* __restrict__ gate_b,
                                                         float* __restrict__ agg,
                                                         int layer) {
    const int lane = threadIdx.x & 31;
    const int wid  = threadIdx.x >> 5;
    const int half = lane >> 4;           // edge within wave
    const int ln   = lane & 15;           // feature group (4 floats)
    const int e    = blockIdx.x * 16 + wid * 2 + half;

    const int r = ei[e];        // source
    const int c = ei[NE + e];   // target
    const float* gw = gate_w + layer * 2 * HID;

    float4 hr = *(const float4*)(h + (size_t)r * HID + 4 * ln);
    float4 hc = *(const float4*)(h + (size_t)c * HID + 4 * ln);
    float4 wa = *(const float4*)(gw + 4 * ln);
    float4 wb = *(const float4*)(gw + HID + 4 * ln);

    float p = hr.x * wa.x + hr.y * wa.y + hr.z * wa.z + hr.w * wa.w
            + hc.x * wb.x + hc.y * wb.y + hc.z * wb.z + hc.w * wb.w;
#pragma unroll
    for (int off = 8; off >= 1; off >>= 1) p += __shfl_xor(p, off, 16);

    const float g    = tanhf(p + gate_b[layer]);
    const float norm = g * nd[r] * nd[c];

    float* dst = agg + (size_t)c * HID + 4 * ln;
    atomicAdd(dst + 0, norm * (hc.x + hr.x));
    atomicAdd(dst + 1, norm * (hc.y + hr.y));
    atomicAdd(dst + 2, norm * (hc.z + hr.z));
    atomicAdd(dst + 3, norm * (hc.w + hr.w));
}

// ---------------------------------------------------------------- h = EPS*raw + agg/cnt (12500 blocks exact)
__global__ void combine_kernel(float* __restrict__ h, const float* __restrict__ raw,
                               const float* __restrict__ agg, const float* __restrict__ cnt) {
    int i = blockIdx.x * blockDim.x + threadIdx.x;
    h[i] = EPSF * raw[i] + agg[i] / cnt[i >> 6];   // HID == 64
}

// ---------------------------------------------------------------- logits = h @ W2^T + b2
// W2 staged to LDS as f16, zero-padded to 48 columns -> unconditional B loads.
__global__ __launch_bounds__(256) void out_wmma(const float* __restrict__ h,
                                                const float* __restrict__ W2,
                                                const float* __restrict__ b2,
                                                float* __restrict__ out) {
    __shared__ alignas(32) _Float16 sW2[NPAD * HID];   // 6 KB, [48][64]

    for (int i = threadIdx.x; i < (NPAD * HID) / 8; i += 256) {
        const int base = i * 8;
        const int n = base >> 6;           // row of padded W2
        v8h v = {};
        if (n < NCLS) {
            const float* src = W2 + base;  // same flat layout for n < 40
            v = pack8(*(const float4*)(src), *(const float4*)(src + 4));
        }
        *(v8h*)(sW2 + base) = v;
    }
    __syncthreads();

    const int lane    = threadIdx.x & 31;
    const int wid     = threadIdx.x >> 5;
    const int rowBase = blockIdx.x * 128 + wid * 16;
    if (rowBase >= NN) return;
    const int laneLo = lane & 15;
    const int laneHi = lane >> 4;

    const float* arow = h + (size_t)(rowBase + laneLo) * HID;
    v8f acc[3] = {};

    for (int k0 = 0; k0 < HID; k0 += 32) {
        const int kA = k0 + laneHi * 16;
        v16h A = pack16(*(const float4*)(arow + kA),
                        *(const float4*)(arow + kA + 4),
                        *(const float4*)(arow + kA + 8),
                        *(const float4*)(arow + kA + 12));
        const int kB = k0 + laneHi * 16;
#pragma unroll
        for (int t = 0; t < 3; ++t) {
            v16h B = *(const v16h*)(sW2 + (t * 16 + laneLo) * HID + kB);
            acc[t] = __builtin_amdgcn_wmma_f32_16x16x32_f16(
                false, A, false, B, (short)0, acc[t], false, false);
        }
    }

#pragma unroll
    for (int t = 0; t < 3; ++t) {
        const int n = t * 16 + laneLo;
        if (n >= NCLS) continue;
        const float bias = b2[n];
#pragma unroll
        for (int v = 0; v < 8; ++v) {
            const int m = rowBase + v + 8 * laneHi;
            out[(size_t)m * NCLS + n] = acc[t][v] + bias;
        }
    }
}

// ---------------------------------------------------------------- row-wise log_softmax (6250 blocks exact)
__global__ __launch_bounds__(256) void logsoftmax_kernel(float* __restrict__ out) {
    const int lane = threadIdx.x & 31;
    const int wid  = threadIdx.x >> 5;
    const int node = blockIdx.x * 8 + wid;
    float* rowp = out + (size_t)node * NCLS;

    float v0 = (lane < NCLS)      ? rowp[lane]      : -INFINITY;
    float v1 = (lane + 32 < NCLS) ? rowp[lane + 32] : -INFINITY;

    float mx = fmaxf(v0, v1);
#pragma unroll
    for (int off = 16; off >= 1; off >>= 1) mx = fmaxf(mx, __shfl_xor(mx, off, 32));

    float s = ((lane < NCLS) ? __expf(v0 - mx) : 0.0f) +
              ((lane + 32 < NCLS) ? __expf(v1 - mx) : 0.0f);
#pragma unroll
    for (int off = 16; off >= 1; off >>= 1) s += __shfl_xor(s, off, 32);

    const float lse = mx + logf(s);
    if (lane < NCLS)      rowp[lane]      = v0 - lse;
    if (lane + 32 < NCLS) rowp[lane + 32] = v1 - lse;
}

// ---------------------------------------------------------------- launch
extern "C" void kernel_launch(void* const* d_in, const int* in_sizes, int n_in,
                              void* d_out, int out_size, void* d_ws, size_t ws_size,
                              hipStream_t stream) {
    const float* x      = (const float*)d_in[0];
    const int*   ei     = (const int*)  d_in[1];   // [2, NE] flat (int32 per harness convention)
    const float* W1     = (const float*)d_in[2];
    const float* b1     = (const float*)d_in[3];
    const float* W2     = (const float*)d_in[4];
    const float* b2     = (const float*)d_in[5];
    const float* gate_w = (const float*)d_in[6];
    const float* gate_b = (const float*)d_in[7];
    float* out = (float*)d_out;

    // workspace carve-out (all f32): raw | h | agg | nd | cnt  => ~38.8 MB
    float* raw = (float*)d_ws;
    float* h   = raw + NH;
    float* agg = h   + NH;
    float* nd  = agg + NH;
    float* cnt = nd  + NN;

    // degrees / norms (zeroed every call -> deterministic under graph replay)
    fill_zero<<<512, 256, 0, stream>>>(nd, 2 * NN);            // nd and cnt are contiguous
    degree_kernel<<<NE / 256, 256, 0, stream>>>(ei, nd, cnt);
    node_norm<<<(NN + 255) / 256, 256, 0, stream>>>(nd, cnt);

    // input projection (WMMA f16 -> f32, LDS-staged B)
    proj_wmma<<<(3125 + 7) / 8, 256, 0, stream>>>(x, W1, b1, raw, h);

    // two gated message-passing layers
    for (int l = 0; l < 2; ++l) {
        fill_zero<<<2048, 256, 0, stream>>>(agg, NH);
        edge_gate_scatter<<<NE / 16, 256, 0, stream>>>(h, ei, nd, gate_w, gate_b, agg, l);
        combine_kernel<<<NH / 256, 256, 0, stream>>>(h, raw, agg, cnt);
    }

    // classifier (WMMA, LDS-staged zero-padded W2) + log-softmax
    out_wmma<<<(3125 + 7) / 8, 256, 0, stream>>>(h, W2, b2, out);
    logsoftmax_kernel<<<NN / 8, 256, 0, stream>>>(out);
}
